// SelfAttentionHead_37658273251593
// MI455X (gfx1250) — matrix operand order repaired and checked by reading, not compiled, not run on hardware
//
#include <hip/hip_runtime.h>

#define B_ 8
#define T_ 2048
#define E_ 1024
#define H_ 64

typedef __attribute__((ext_vector_type(16))) __bf16   v16bf;
typedef __attribute__((ext_vector_type(8)))  float    v8f;
typedef __attribute__((ext_vector_type(4)))  float    f4;
typedef __attribute__((ext_vector_type(4)))  unsigned u4;

union FragBF { v16bf v; u4 q[2]; unsigned u[8]; unsigned short s[16]; };

// Two fp32 -> packed bf16 pair (RNE) in one VALU op (gfx12-family opcode).
__device__ inline unsigned cvtpk_bf16(float lo, float hi) {
    unsigned r;
    asm("v_cvt_pk_bf16_f32 %0, %1, %2" : "=v"(r) : "v"(lo), "v"(hi));
    return r;
}
__device__ inline unsigned short f2bf(float f) {
    return (unsigned short)cvtpk_bf16(f, f);
}
__device__ inline v8f wmma_bf16(v16bf a, v16bf b, v8f c) {
    // D = A(16x32 bf16) * B(32x16 bf16) + C(16x16 f32)
    return __builtin_amdgcn_wmma_f32_16x16x32_bf16(false, a, false, b, (short)0, c, false, false);
}
__device__ inline v8f vzero() {
    v8f z;
#pragma unroll
    for (int i = 0; i < 8; ++i) z[i] = 0.f;
    return z;
}

// ---------------- Kernel 0: weights fp32 [E][H] -> bf16 transposed [H][E] ----
__global__ __launch_bounds__(256) void cvtW(const float* __restrict__ Wk,
                                            const float* __restrict__ Wq,
                                            const float* __restrict__ Wv,
                                            unsigned short* __restrict__ Wtk,
                                            unsigned short* __restrict__ Wtq,
                                            unsigned short* __restrict__ Wtv) {
    int idx = blockIdx.x * 256 + threadIdx.x;   // 3 * 64 * 1024 total
    int w   = idx >> 16;                        // which weight
    int r   = idx & 65535;                      // r = h*1024 + e
    int h   = r >> 10;
    int e   = r & 1023;
    const float* W = (w == 0) ? Wk : (w == 1) ? Wq : Wv;
    unsigned short* Wt = (w == 0) ? Wtk : (w == 1) ? Wtq : Wtv;
    Wt[r] = f2bf(W[e * H_ + h]);
}

// ---------------- Kernel 1: QKV projection (bf16 WMMA, fp32 accum) ----------
// One wave owns a 16-row tile of M = B*T; 12 C-tiles (Q,K,V x 4 col-tiles).
__global__ __launch_bounds__(256) void qkv_proj(const float* __restrict__ x,
                                                const unsigned short* __restrict__ Wtk,
                                                const unsigned short* __restrict__ Wtq,
                                                const unsigned short* __restrict__ Wtv,
                                                unsigned short* __restrict__ Qo,
                                                unsigned short* __restrict__ Ko,
                                                unsigned short* __restrict__ Vt) {
    const int lane = threadIdx.x & 31;
    const int w    = threadIdx.x >> 5;
    const int h    = lane >> 4;      // lane half (A/B layout selector)
    const int ln   = lane & 15;
    const int rowbase = blockIdx.x * 128 + w * 16;   // [0, 16384)
    const int row     = rowbase + ln;

    v8f cq[4], ck[4], cv[4];
#pragma unroll
    for (int nt = 0; nt < 4; ++nt) { cq[nt] = vzero(); ck[nt] = vzero(); cv[nt] = vzero(); }

    const float* xr = x + (size_t)row * E_;

    for (int kt = 0; kt < E_ / 32; ++kt) {
        const int kb = kt * 32;
        // A fragment: x[row][kb..kb+31], contiguous K order (K = 16h + s)
        FragBF a;
        f4 c0 = *(const f4*)(xr + kb + 16 * h);
        f4 c1 = *(const f4*)(xr + kb + 16 * h + 4);
        f4 c2 = *(const f4*)(xr + kb + 16 * h + 8);
        f4 c3 = *(const f4*)(xr + kb + 16 * h + 12);
        a.u[0] = cvtpk_bf16(c0[0], c0[1]); a.u[1] = cvtpk_bf16(c0[2], c0[3]);
        a.u[2] = cvtpk_bf16(c1[0], c1[1]); a.u[3] = cvtpk_bf16(c1[2], c1[3]);
        a.u[4] = cvtpk_bf16(c2[0], c2[1]); a.u[5] = cvtpk_bf16(c2[2], c2[3]);
        a.u[6] = cvtpk_bf16(c3[0], c3[1]); a.u[7] = cvtpk_bf16(c3[2], c3[3]);

#pragma unroll
        for (int nt = 0; nt < 4; ++nt) {
            const int n = nt * 16 + ln;   // output column (B-layout: N = lane%16)
            FragBF bq, bk, bv;
            const u4* pq = (const u4*)(Wtq + (size_t)n * E_ + kb + 16 * h);
            bq.q[0] = pq[0]; bq.q[1] = pq[1];
            const u4* pk = (const u4*)(Wtk + (size_t)n * E_ + kb + 16 * h);
            bk.q[0] = pk[0]; bk.q[1] = pk[1];
            const u4* pv = (const u4*)(Wtv + (size_t)n * E_ + kb + 16 * h);
            bv.q[0] = pv[0]; bv.q[1] = pv[1];
            cq[nt] = wmma_bf16(a.v, bq.v, cq[nt]);
            ck[nt] = wmma_bf16(a.v, bk.v, ck[nt]);
            cv[nt] = wmma_bf16(a.v, bv.v, cv[nt]);
        }
    }

    const int b  = rowbase >> 11;      // batch
    const int tb = rowbase & 2047;     // token within batch
#pragma unroll
    for (int nt = 0; nt < 4; ++nt) {
        // C-layout: row = r + 8h, col = nt*16 + ln. Pack row pairs with one cvt.
#pragma unroll
        for (int r = 0; r < 8; r += 2) {
            const size_t a0 = (size_t)(rowbase + r + 8 * h) * H_ + nt * 16 + ln;
            unsigned pq = cvtpk_bf16(cq[nt][r], cq[nt][r + 1]);
            Qo[a0]      = (unsigned short)pq;
            Qo[a0 + H_] = (unsigned short)(pq >> 16);   // d16_hi store
            unsigned pk = cvtpk_bf16(ck[nt][r], ck[nt][r + 1]);
            Ko[a0]      = (unsigned short)pk;
            Ko[a0 + H_] = (unsigned short)(pk >> 16);
        }
        // V transposed store: per lane, 8 consecutive tokens -> one 16B store
        u4 pv;
        pv[0] = cvtpk_bf16(cv[nt][0], cv[nt][1]);
        pv[1] = cvtpk_bf16(cv[nt][2], cv[nt][3]);
        pv[2] = cvtpk_bf16(cv[nt][4], cv[nt][5]);
        pv[3] = cvtpk_bf16(cv[nt][6], cv[nt][7]);
        *(u4*)(Vt + (size_t)b * H_ * T_ + (size_t)(nt * 16 + ln) * T_ + tb + 8 * h) = pv;
    }
}

// ---------------- Kernel 2: causal flash attention (bf16 WMMA) --------------
// One wave per 16-query tile; 32 keys per iteration; fp32 softmax state.
__global__ __launch_bounds__(256) void attn(const unsigned short* __restrict__ Q,
                                            const unsigned short* __restrict__ K,
                                            const unsigned short* __restrict__ Vt,
                                            float* __restrict__ out) {
    __shared__ __align__(16) unsigned short pl[8][16 * 40];  // per-wave P tile, 80B row stride
    const int lane = threadIdx.x & 31;
    const int w    = threadIdx.x >> 5;
    const int h    = lane >> 4;
    const int ln   = lane & 15;
    const int tile  = blockIdx.x * 8 + w;   // 1024 q-tiles total
    const int b     = tile >> 7;
    const int qt    = tile & 127;
    const int qbase = qt * 16;

    const unsigned short* Qb = Q  + ((size_t)b * T_ + qbase) * H_;
    const unsigned short* Kb = K  + (size_t)b * T_ * H_;
    const unsigned short* Vb = Vt + (size_t)b * H_ * T_;

    // Q A-fragments (dims 0..31 and 32..63), loaded once
    FragBF qa0, qa1;
    qa0.q[0] = *(const u4*)(Qb + (size_t)ln * H_ + 16 * h);
    qa0.q[1] = *(const u4*)(Qb + (size_t)ln * H_ + 8 + 16 * h);
    qa1.q[0] = *(const u4*)(Qb + (size_t)ln * H_ + 32 + 16 * h);
    qa1.q[1] = *(const u4*)(Qb + (size_t)ln * H_ + 40 + 16 * h);

    v8f o[4];
    float m[8], l[8];
#pragma unroll
    for (int nt = 0; nt < 4; ++nt) o[nt] = vzero();
#pragma unroll
    for (int r = 0; r < 8; ++r) { m[r] = -1e30f; l[r] = 0.f; }

    const float scale = 0.03125f;               // E^-0.5 = 1/32
    const float L2E   = 1.4426950408889634f;
    const int   nkt   = (qbase + 15) / 32 + 1;  // key tiles (triangular)

    unsigned short* lds = &pl[w][0];

    for (int kt = 0; kt < nkt; ++kt) {
        const int k0 = kt * 32;
        if (k0 + 63 < T_)  // prefetch next key tile (global_prefetch_b8)
            __builtin_prefetch(Kb + (size_t)(k0 + 32 + ln + 16 * h) * H_, 0, 1);

        // S = Q * K^T  (two 16x16 tiles: keys k0..k0+15 and k0+16..k0+31)
        v8f s0 = vzero(), s1 = vzero();
        {
            FragBF b00, b01, b10, b11;
            const u4* p0 = (const u4*)(Kb + (size_t)(k0 + ln) * H_ + 16 * h);
            b00.q[0] = p0[0]; b00.q[1] = p0[1];
            const u4* p1 = (const u4*)(Kb + (size_t)(k0 + ln) * H_ + 32 + 16 * h);
            b01.q[0] = p1[0]; b01.q[1] = p1[1];
            const u4* p2 = (const u4*)(Kb + (size_t)(k0 + 16 + ln) * H_ + 16 * h);
            b10.q[0] = p2[0]; b10.q[1] = p2[1];
            const u4* p3 = (const u4*)(Kb + (size_t)(k0 + 16 + ln) * H_ + 32 + 16 * h);
            b11.q[0] = p3[0]; b11.q[1] = p3[1];
            s0 = wmma_bf16(qa0.v, b00.v, s0);
            s0 = wmma_bf16(qa1.v, b01.v, s0);
            s1 = wmma_bf16(qa0.v, b10.v, s1);
            s1 = wmma_bf16(qa1.v, b11.v, s1);
        }

        // online softmax over 32 keys (row lives on lanes h*16..h*16+15)
        const int key0 = k0 + ln, key1 = k0 + 16 + ln;
        float p0[8], p1[8], fr[8];
#pragma unroll
        for (int r = 0; r < 8; ++r) {
            const int qrow = qbase + r + 8 * h;
            float a0 = s0[r] * scale, a1 = s1[r] * scale;
            a0 = (key0 <= qrow) ? a0 : -1e30f;
            a1 = (key1 <= qrow) ? a1 : -1e30f;
            float rm = fmaxf(a0, a1);
            rm = fmaxf(rm, __shfl_xor(rm, 1, 32));
            rm = fmaxf(rm, __shfl_xor(rm, 2, 32));
            rm = fmaxf(rm, __shfl_xor(rm, 4, 32));
            rm = fmaxf(rm, __shfl_xor(rm, 8, 32));
            float mn = fmaxf(m[r], rm);
            float f  = __builtin_amdgcn_exp2f((m[r] - mn) * L2E);
            p0[r] = __builtin_amdgcn_exp2f((a0 - mn) * L2E);
            p1[r] = __builtin_amdgcn_exp2f((a1 - mn) * L2E);
            float rs = p0[r] + p1[r];
            rs += __shfl_xor(rs, 1, 32);
            rs += __shfl_xor(rs, 2, 32);
            rs += __shfl_xor(rs, 4, 32);
            rs += __shfl_xor(rs, 8, 32);
            l[r] = l[r] * f + rs;
            m[r] = mn;
            fr[r] = f;
        }
#pragma unroll
        for (int nt = 0; nt < 4; ++nt)
#pragma unroll
            for (int r = 0; r < 8; ++r) o[nt][r] *= fr[r];

        // P: C-layout -> A-layout restripe through per-wave LDS tile
#pragma unroll
        for (int r = 0; r < 8; ++r) {
            const int rr = r + 8 * h;
            unsigned pp = cvtpk_bf16(p0[r], p1[r]);   // lo: key tile 0, hi: key tile 1
            lds[rr * 40 + ln]      = (unsigned short)pp;
            lds[rr * 40 + 16 + ln] = (unsigned short)(pp >> 16);
        }
        FragBF pa;
        pa.q[0] = *(const u4*)(lds + ln * 40 + 16 * h);
        pa.q[1] = *(const u4*)(lds + ln * 40 + 8 + 16 * h);

        // O += P * V   (V^T storage makes B-fragments contiguous)
#pragma unroll
        for (int nt = 0; nt < 4; ++nt) {
            FragBF vb;
            const u4* pv = (const u4*)(Vb + (size_t)(nt * 16 + ln) * T_ + k0 + 16 * h);
            vb.q[0] = pv[0]; vb.q[1] = pv[1];
            o[nt] = wmma_bf16(pa.v, vb.v, o[nt]);
        }
    }

    // epilogue: normalize and store fp32
#pragma unroll
    for (int r = 0; r < 8; ++r) l[r] = 1.0f / l[r];
    float* ob = out + ((size_t)b * T_ + qbase) * H_;
#pragma unroll
    for (int nt = 0; nt < 4; ++nt)
#pragma unroll
        for (int r = 0; r < 8; ++r)
            ob[(size_t)(r + 8 * h) * H_ + nt * 16 + ln] = o[nt][r] * l[r];
}

extern "C" void kernel_launch(void* const* d_in, const int* in_sizes, int n_in,
                              void* d_out, int out_size, void* d_ws, size_t ws_size,
                              hipStream_t stream) {
    (void)in_sizes; (void)n_in; (void)out_size; (void)ws_size;
    const float* x  = (const float*)d_in[0];
    const float* Wk = (const float*)d_in[1];
    const float* Wq = (const float*)d_in[2];
    const float* Wv = (const float*)d_in[3];
    float* out = (float*)d_out;

    char* ws = (char*)d_ws;
    unsigned short* Wtk = (unsigned short*)(ws);
    unsigned short* Wtq = (unsigned short*)(ws + (size_t)131072);
    unsigned short* Wtv = (unsigned short*)(ws + (size_t)262144);
    unsigned short* Qb  = (unsigned short*)(ws + (size_t)393216);
    unsigned short* Kb  = (unsigned short*)(ws + (size_t)393216 + (size_t)2097152);
    unsigned short* Vt  = (unsigned short*)(ws + (size_t)393216 + (size_t)2 * 2097152);

    cvtW<<<768, 256, 0, stream>>>(Wk, Wq, Wv, Wtk, Wtq, Wtv);           // 3*64K elems
    qkv_proj<<<128, 256, 0, stream>>>(x, Wtk, Wtq, Wtv, Qb, Kb, Vt);    // M=16384 / 128 rows per WG
    attn<<<128, 256, 0, stream>>>(Qb, Kb, Vt, out);                     // 1024 q-tiles / 8 per WG
}
